// SwinTransformerBlock_2001454760398
// MI455X (gfx1250) — matrix operand order repaired and checked by reading, not compiled, not run on hardware
//
#include <hip/hip_runtime.h>
#include <math.h>

typedef __attribute__((ext_vector_type(16))) _Float16 v16h;
typedef __attribute__((ext_vector_type(8)))  float    v8f;

// 16-byte POD for b128 moves (avoid HIP vector-type ctors inside unions)
struct alignas(16) B16 { unsigned long long lo, hi; };

__device__ __forceinline__ v8f v8zero() {
    v8f z;
#pragma unroll
    for (int i = 0; i < 8; ++i) z[i] = 0.0f;
    return z;
}

// Wave-local LDS write->read ordering (DS ops are in-order per wave on CDNA5;
// we only need the data-return wait + a compiler fence).
__device__ __forceinline__ void wave_lds_sync() {
    __builtin_amdgcn_wave_barrier();
    asm volatile("s_wait_dscnt 0x0" ::: "memory");
    __builtin_amdgcn_wave_barrier();
}

// CDNA5 async global->LDS copy (ASYNCcnt-tracked, no VGPR staging).
// LDS destination address is lds_off; generic LDS pointer low 32 bits == LDS offset.
__device__ __forceinline__ void async_ld_b128(const _Float16* lds_dst, const _Float16* gsrc) {
    unsigned lds_off = (unsigned)(unsigned long long)lds_dst;
    asm volatile("global_load_async_to_lds_b128 %0, %1, off"
                 :: "v"(lds_off), "v"(gsrc) : "memory");
}
__device__ __forceinline__ void wait_async0() {
    asm volatile("s_wait_asynccnt 0x0" ::: "memory");
}

// ---------------------------------------------------------------- constants
constexpr int DIM   = 384;
constexpr int HEADS = 12;
constexpr int HD    = 32;
constexpr int NWIN  = 343;   // 7^3 tokens per window
constexpr int NP    = 352;   // padded to 11*32
constexpr int NW    = 8;     // windows per batch (2x2x2)
constexpr int BATCH = 8;
constexpr int L     = 2744;  // 14^3
constexpr int TOK   = BATCH * NW * NWIN; // 21952 = 64*343 (exact 64-multiples)
constexpr int HID   = 4 * DIM;           // 1536

// ---------------------------------------------------------------- helpers
__device__ __forceinline__ int cat14(int g) { return (g < 7) ? 0 : ((g < 11) ? 1 : 2); }

__device__ __forceinline__ float cpb_coord(int i) {
    float t = (float)(i - 6) * (8.0f / 6.0f);
    float v = log2f(fabsf(t) + 1.0f) * (1.0f / 3.0f);
    return (t < 0.0f) ? -v : v;
}

// =================================================================== kernels

// Roll(-3) + window partition + f32->f16: xw[(win*343+n)*384+c]
__global__ __launch_bounds__(256)
void gather_xw_kernel(const float* __restrict__ x, _Float16* __restrict__ xw) {
    int e = blockIdx.x * 256 + threadIdx.x;
    if (e >= TOK * DIM) return;
    int c = e % DIM, row = e / DIM;
    int win = row / NWIN, n = row % NWIN;
    int b = win >> 3, widx = win & 7;
    int wz = (widx >> 2) & 1, wh = (widx >> 1) & 1, ww = widx & 1;
    int nz = n / 49, rem = n % 49, nh = rem / 7, nw = rem % 7;
    int gz = (wz * 7 + nz + 3) % 14;
    int gh = (wh * 7 + nh + 3) % 14;
    int gw = (ww * 7 + nw + 3) % 14;
    xw[e] = (_Float16)x[((size_t)b * L + gz * 196 + gh * 14 + gw) * DIM + c];
}

__global__ __launch_bounds__(256)
void cvt_f32_f16_kernel(const float* __restrict__ s, _Float16* __restrict__ d, int n) {
    int i = blockIdx.x * 256 + threadIdx.x;
    if (i < n) d[i] = (_Float16)s[i];
}

__global__ __launch_bounds__(256)
void build_qkv_bias_kernel(const float* __restrict__ qb, const float* __restrict__ vb,
                           float* __restrict__ out) {
    int i = blockIdx.x * 256 + threadIdx.x;
    if (i >= 3 * DIM) return;
    out[i] = (i < DIM) ? qb[i] : ((i < 2 * DIM) ? 0.0f : vb[i - 2 * DIM]);
}

// CPB MLP table: tbl[row][h] = (relu(coords @ w1^T + b1) @ w2^T)[h], row in [0,13^3)
__global__ __launch_bounds__(128)
void cpb_kernel(const float* __restrict__ w1, const float* __restrict__ b1,
                const float* __restrict__ w2, float* __restrict__ tbl) {
    int row = blockIdx.x;
    int i0 = row / 169, i1 = (row / 13) % 13, i2 = row % 13;
    float c0 = cpb_coord(i0), c1 = cpb_coord(i1), c2 = cpb_coord(i2);
    float acc[HEADS];
#pragma unroll
    for (int h = 0; h < HEADS; ++h) acc[h] = 0.0f;
    for (int j = threadIdx.x; j < 512; j += 128) {
        float hv = c0 * w1[j * 3] + c1 * w1[j * 3 + 1] + c2 * w1[j * 3 + 2] + b1[j];
        hv = fmaxf(hv, 0.0f);
#pragma unroll
        for (int h = 0; h < HEADS; ++h) acc[h] += hv * w2[h * 512 + j];
    }
    __shared__ float red[128];
    for (int h = 0; h < HEADS; ++h) {
        red[threadIdx.x] = acc[h];
        __syncthreads();
        for (int st = 64; st > 0; st >>= 1) {
            if (threadIdx.x < st) red[threadIdx.x] += red[threadIdx.x + st];
            __syncthreads();
        }
        if (threadIdx.x == 0) tbl[row * HEADS + h] = red[0];
        __syncthreads();
    }
}

// Generic WMMA GEMM: out[M,N] = A[M,K] @ W[N,K]^T + bias ; act 0=none 1=gelu(exact)
// 4 waves, 64x64 tile, K-step 32, double-buffered async global->LDS copies.
// All B fragments are preloaded so the 4 WMMAs issue back-to-back with partial
// DScnt waits instead of a full drain per WMMA.
__global__ __launch_bounds__(128)
void gemm_wmma_kernel(const _Float16* __restrict__ A, const _Float16* __restrict__ W,
                      const float* __restrict__ bias, float* __restrict__ out32,
                      _Float16* __restrict__ out16, int M, int K, int Nn, int act) {
    constexpr int LDT = 40; // padded halves per row (80B, 16B-aligned chunks)
    __shared__ alignas(16) _Float16 At[2][64 * LDT];
    __shared__ alignas(16) _Float16 Bt[2][64 * LDT];
    const int tid = threadIdx.x, lane = tid & 31, wave = tid >> 5;
    const int m0 = blockIdx.x * 64, n0 = blockIdx.y * 64;

    v8f acc[4];
#pragma unroll
    for (int i = 0; i < 4; ++i) acc[i] = v8zero();

    // 64 rows x 32 halves per matrix = 256 b128 chunks, 2 per thread per matrix
    auto prefetch = [&](int kk, int buf) {
#pragma unroll
        for (int c0 = 0; c0 < 2; ++c0) {
            int c = tid + c0 * 128;
            int r = c >> 2, cb = (c & 3) * 8;
            async_ld_b128(&At[buf][r * LDT + cb], &A[(size_t)(m0 + r) * K + kk + cb]);
            async_ld_b128(&Bt[buf][r * LDT + cb], &W[(size_t)(n0 + r) * K + kk + cb]);
        }
    };

    const int nsteps = K / 32;
    prefetch(0, 0);
    for (int s = 0; s < nsteps; ++s) {
        const int buf = s & 1;
        wait_async0();          // this wave's copies into `buf` are complete
        __syncthreads();        // all waves' copies complete; prev buffer free
        if (s + 1 < nsteps) prefetch((s + 1) * 32, buf ^ 1); // overlap with compute
        v16h a;
        {
            int lr = lane & 15, kh = (lane >> 4) * 16;
            const _Float16* p = &At[buf][(wave * 16 + lr) * LDT];
            ((B16*)&a)[0] = *(const B16*)(p + kh);
            ((B16*)&a)[1] = *(const B16*)(p + 8 + kh);
        }
        v16h bfr[4];
#pragma unroll
        for (int nb = 0; nb < 4; ++nb) {    // batch all fragment loads first
            int nc = nb * 16 + (lane & 15), kb = (lane >> 4) * 16;
            const _Float16* p = &Bt[buf][nc * LDT + kb];
            ((B16*)&bfr[nb])[0] = *(const B16*)(p);
            ((B16*)&bfr[nb])[1] = *(const B16*)(p + 8);
        }
#pragma unroll
        for (int nb = 0; nb < 4; ++nb) {    // then run the WMMA chain
            acc[nb] = __builtin_amdgcn_wmma_f32_16x16x32_f16(
                false, a, false, bfr[nb], (short)0, acc[nb], false, false);
        }
    }
    const int colb = n0 + (lane & 15);
    const int rs = (lane < 16) ? 0 : 8;
#pragma unroll
    for (int nb = 0; nb < 4; ++nb) {
        int col = colb + nb * 16;
        float bv = bias ? bias[col] : 0.0f;
#pragma unroll
        for (int r = 0; r < 8; ++r) {
            int row = m0 + wave * 16 + rs + r;
            float v = acc[nb][r] + bv;
            if (act == 1) v = 0.5f * v * (1.0f + erff(v * 0.70710678118654752f));
            size_t o = (size_t)row * Nn + col;
            if (out32) out32[o] = v;
            if (out16) out16[o] = (_Float16)v;
        }
    }
}

// Flash attention per (window, head): cosine-sim attn + CPB bias + shift mask.
__global__ __launch_bounds__(256)
void attn_kernel(const float* __restrict__ qkv, const float* __restrict__ logit_scale,
                 const float* __restrict__ tbl, _Float16* __restrict__ out16) {
    __shared__ alignas(16) _Float16 Qn[NP * HD];
    __shared__ alignas(16) _Float16 Kn[NP * HD];
    __shared__ alignas(16) _Float16 Vt[HD * NP];
    __shared__ alignas(16) float    Sb[8][16 * 32];
    __shared__ alignas(16) _Float16 Pb[8][16 * 32];
    __shared__ float Fac[8][16];
    __shared__ float Lrw[8][16];

    const int blk = blockIdx.x;
    const int win = blk / HEADS, head = blk % HEADS;
    const int widx = win & 7;
    const int wz = (widx >> 2) & 1, wh = (widx >> 1) & 1, ww = widx & 1;
    const int tid = threadIdx.x, lane = tid & 31, wave = tid >> 5;

    const float scale = __expf(fminf(logit_scale[head], 4.6051702f)); // log(100)

    // cooperative load + normalize (scale folded into q), v transposed
    for (int row = tid; row < NP; row += 256) {
        if (row < NWIN) {
            const float* base = qkv + (size_t)(win * NWIN + row) * (3 * DIM) + head * HD;
            float q[HD], k[HD], v[HD], qs = 0.0f, ks = 0.0f;
#pragma unroll
            for (int d = 0; d < HD; ++d) {
                q[d] = base[d]; k[d] = base[DIM + d]; v[d] = base[2 * DIM + d];
                qs += q[d] * q[d]; ks += k[d] * k[d];
            }
            float qi = scale / fmaxf(sqrtf(qs), 1e-12f);
            float ki = 1.0f / fmaxf(sqrtf(ks), 1e-12f);
#pragma unroll
            for (int d = 0; d < HD; ++d) {
                Qn[row * HD + d] = (_Float16)(q[d] * qi);
                Kn[row * HD + d] = (_Float16)(k[d] * ki);
                Vt[d * NP + row] = (_Float16)v[d];
            }
        } else {
#pragma unroll
            for (int d = 0; d < HD; ++d) {
                Qn[row * HD + d] = (_Float16)0.0f;
                Kn[row * HD + d] = (_Float16)0.0f;
                Vt[d * NP + row] = (_Float16)0.0f;
            }
        }
    }
    __syncthreads();

    for (int qt = wave; qt < NP / 16; qt += 8) {   // wave-local loop, no block syncs
        const int q0 = qt * 16;
        v16h aq;
        {
            int lr = lane & 15, kh = (lane >> 4) * 16;
            const _Float16* p = &Qn[(q0 + lr) * HD];
            ((B16*)&aq)[0] = *(const B16*)(p + kh);
            ((B16*)&aq)[1] = *(const B16*)(p + 8 + kh);
        }
        v8f o0 = v8zero(), o1 = v8zero();
        float mrow = -1e30f, lrow = 0.0f;          // lanes 0..15: own local row = lane
        const int nrow = q0 + lane;
        const bool nvalid = (lane < 16) && (nrow < NWIN);
        int nz = 0, nh = 0, nw = 0, cntn = 0;
        if (nvalid) {
            nz = nrow / 49; int rem = nrow % 49; nh = rem / 7; nw = rem % 7;
            cntn = cat14(wz * 7 + nz) * 9 + cat14(wh * 7 + nh) * 3 + cat14(ww * 7 + nw);
        }

        for (int j = 0; j < NP / 32; ++j) {
            const int kb0 = j * 32;
            v8f s0 = v8zero(), s1 = v8zero();
            {
                v16h b0, b1;
                int nc = kb0 + (lane & 15), kb = (lane >> 4) * 16;
                const _Float16* p = &Kn[nc * HD + kb];
                ((B16*)&b0)[0] = *(const B16*)(p);
                ((B16*)&b0)[1] = *(const B16*)(p + 8);
                p = &Kn[(nc + 16) * HD + kb];
                ((B16*)&b1)[0] = *(const B16*)(p);
                ((B16*)&b1)[1] = *(const B16*)(p + 8);
                s0 = __builtin_amdgcn_wmma_f32_16x16x32_f16(false, aq, false, b0, (short)0, s0, false, false);
                s1 = __builtin_amdgcn_wmma_f32_16x16x32_f16(false, aq, false, b1, (short)0, s1, false, false);
            }
            {   // scatter score tiles to this wave's LDS strip
                int col = lane & 15, rs = (lane < 16) ? 0 : 8;
#pragma unroll
                for (int r = 0; r < 8; ++r) {
                    Sb[wave][(rs + r) * 32 + col]      = s0[r];
                    Sb[wave][(rs + r) * 32 + 16 + col] = s1[r];
                }
            }
            wave_lds_sync();
            if (lane < 16) {                        // online softmax, row owner lanes
                float e[32], mnew = mrow;
#pragma unroll
                for (int c = 0; c < 32; ++c) {
                    int mcol = kb0 + c;
                    float s;
                    if (nvalid && mcol < NWIN) {
                        s = Sb[wave][lane * 32 + c];
                        int mz = mcol / 49, rem = mcol % 49, mh = rem / 7, mw = rem % 7;
                        int idx = ((nz - mz + 6) * 13 + (nh - mh + 6)) * 13 + (nw - mw + 6);
                        float t = tbl[(size_t)idx * HEADS + head];
                        s += 16.0f / (1.0f + __expf(-t));
                        int cntm = cat14(wz * 7 + mz) * 9 + cat14(wh * 7 + mh) * 3 + cat14(ww * 7 + mw);
                        if (cntm != cntn) s -= 100.0f;
                    } else s = -1e30f;
                    e[c] = s;
                    mnew = fmaxf(mnew, s);
                }
                float fac = __expf(mrow - mnew), ls = 0.0f;
#pragma unroll
                for (int c = 0; c < 32; ++c) {
                    float pv = (e[c] <= -1e29f) ? 0.0f : __expf(e[c] - mnew);
                    Pb[wave][lane * 32 + c] = (_Float16)pv;
                    ls += pv;
                }
                mrow = mnew;
                lrow = lrow * fac + ls;
                Fac[wave][lane] = fac;
            }
            wave_lds_sync();
            {   // rescale running output, then O += P @ V
                int rs = (lane < 16) ? 0 : 8;
#pragma unroll
                for (int r = 0; r < 8; ++r) {
                    float f = Fac[wave][rs + r];
                    o0[r] *= f; o1[r] *= f;
                }
                v16h ap, bv0, bv1;
                int lr = lane & 15, kh = (lane >> 4) * 16;
                const _Float16* p = &Pb[wave][lr * 32];
                ((B16*)&ap)[0] = *(const B16*)(p + kh);
                ((B16*)&ap)[1] = *(const B16*)(p + 8 + kh);
                int dc = lane & 15, kb = kb0 + (lane >> 4) * 16;
                p = &Vt[dc * NP + kb];
                ((B16*)&bv0)[0] = *(const B16*)(p);
                ((B16*)&bv0)[1] = *(const B16*)(p + 8);
                p = &Vt[(dc + 16) * NP + kb];
                ((B16*)&bv1)[0] = *(const B16*)(p);
                ((B16*)&bv1)[1] = *(const B16*)(p + 8);
                o0 = __builtin_amdgcn_wmma_f32_16x16x32_f16(false, ap, false, bv0, (short)0, o0, false, false);
                o1 = __builtin_amdgcn_wmma_f32_16x16x32_f16(false, ap, false, bv1, (short)0, o1, false, false);
            }
        }
        if (lane < 16) Lrw[wave][lane] = lrow;
        wave_lds_sync();
        {
            int col = lane & 15, rs = (lane < 16) ? 0 : 8;
#pragma unroll
            for (int r = 0; r < 8; ++r) {
                int row = q0 + rs + r;
                if (row < NWIN) {
                    float inv = 1.0f / Lrw[wave][rs + r];
                    size_t base = (size_t)(win * NWIN + row) * DIM + head * HD;
                    out16[base + col]      = (_Float16)(o0[r] * inv);
                    out16[base + 16 + col] = (_Float16)(o1[r] * inv);
                }
            }
        }
    }
}

// out32[t] = LN(src[row(t)]) * g + b + resid[t] ; gathered=1 -> window-reverse+roll
__global__ __launch_bounds__(128)
void ln_res_kernel(const float* __restrict__ src, const float* __restrict__ resid,
                   const float* __restrict__ g, const float* __restrict__ b,
                   float* __restrict__ out32, _Float16* __restrict__ out16, int gathered) {
    const int t = blockIdx.x, tid = threadIdx.x;
    size_t srow;
    if (gathered) {
        int bb = t / L, l = t % L;
        int z = l / 196, h = (l / 14) % 14, w = l % 14;
        int zp = (z + 11) % 14, hp = (h + 11) % 14, wp = (w + 11) % 14;
        int wz = zp / 7, nz = zp % 7, wh = hp / 7, nh = hp % 7, ww = wp / 7, nw = wp % 7;
        int widx = wz * 4 + wh * 2 + ww;
        int n = nz * 49 + nh * 7 + nw;
        srow = (size_t)((bb * NW + widx) * NWIN + n);
    } else srow = (size_t)t;
    const float* p  = src + srow * DIM;
    const float* rp = resid + (size_t)t * DIM;
    __shared__ float s1[128], s2[128];
    float a = 0.0f, a2 = 0.0f;
    for (int c = tid; c < DIM; c += 128) { float v = p[c]; a += v; a2 += v * v; }
    s1[tid] = a; s2[tid] = a2;
    __syncthreads();
    for (int st = 64; st > 0; st >>= 1) {
        if (tid < st) { s1[tid] += s1[tid + st]; s2[tid] += s2[tid + st]; }
        __syncthreads();
    }
    float mean = s1[0] * (1.0f / DIM);
    float var  = s2[0] * (1.0f / DIM) - mean * mean;
    float rs   = rsqrtf(var + 1e-5f);
    for (int c = tid; c < DIM; c += 128) {
        float v = (p[c] - mean) * rs * g[c] + b[c] + rp[c];
        out32[(size_t)t * DIM + c] = v;
        if (out16) out16[(size_t)t * DIM + c] = (_Float16)v;
    }
}

// =================================================================== launch
extern "C" void kernel_launch(void* const* d_in, const int* in_sizes, int n_in,
                              void* d_out, int out_size, void* d_ws, size_t ws_size,
                              hipStream_t stream) {
    const float* x        = (const float*)d_in[0];
    const float* qkv_w    = (const float*)d_in[1];
    const float* q_bias   = (const float*)d_in[2];
    const float* v_bias   = (const float*)d_in[3];
    const float* lscale   = (const float*)d_in[4];
    const float* cpb_w1   = (const float*)d_in[5];
    const float* cpb_b1   = (const float*)d_in[6];
    const float* cpb_w2   = (const float*)d_in[7];
    const float* proj_w   = (const float*)d_in[8];
    const float* proj_b   = (const float*)d_in[9];
    const float* n1g      = (const float*)d_in[10];
    const float* n1b      = (const float*)d_in[11];
    const float* n2g      = (const float*)d_in[12];
    const float* n2b      = (const float*)d_in[13];
    const float* fc1_w    = (const float*)d_in[14];
    const float* fc1_b    = (const float*)d_in[15];
    const float* fc2_w    = (const float*)d_in[16];
    const float* fc2_b    = (const float*)d_in[17];
    float* out = (float*)d_out;

    char* p = (char*)d_ws;
    auto alloc = [&](size_t bytes) -> char* {
        char* r = p;
        p += (bytes + 255) & ~(size_t)255;
        return r;
    };
    _Float16* xw_h    = (_Float16*)alloc((size_t)TOK * DIM * 2);
    _Float16* qkvw_h  = (_Float16*)alloc((size_t)3 * DIM * DIM * 2);
    _Float16* projw_h = (_Float16*)alloc((size_t)DIM * DIM * 2);
    _Float16* fc1w_h  = (_Float16*)alloc((size_t)HID * DIM * 2);
    _Float16* fc2w_h  = (_Float16*)alloc((size_t)DIM * HID * 2);
    float*    qkvb    = (float*)alloc((size_t)3 * DIM * 4);
    float*    tbl     = (float*)alloc((size_t)2197 * HEADS * 4);
    float*    qkv_o   = (float*)alloc((size_t)TOK * 3 * DIM * 4);
    _Float16* attn_h  = (_Float16*)alloc((size_t)TOK * DIM * 2);
    float*    proj_o  = (float*)alloc((size_t)TOK * DIM * 4);
    float*    x1_f    = (float*)alloc((size_t)TOK * DIM * 4);
    _Float16* x1_h    = (_Float16*)alloc((size_t)TOK * DIM * 2);
    _Float16* h_h     = (_Float16*)alloc((size_t)TOK * HID * 2);
    float*    y_f     = (float*)alloc((size_t)TOK * DIM * 4);

    // prep
    build_qkv_bias_kernel<<<(3 * DIM + 255) / 256, 256, 0, stream>>>(q_bias, v_bias, qkvb);
    cvt_f32_f16_kernel<<<(3 * DIM * DIM + 255) / 256, 256, 0, stream>>>(qkv_w, qkvw_h, 3 * DIM * DIM);
    cvt_f32_f16_kernel<<<(DIM * DIM + 255) / 256, 256, 0, stream>>>(proj_w, projw_h, DIM * DIM);
    cvt_f32_f16_kernel<<<(HID * DIM + 255) / 256, 256, 0, stream>>>(fc1_w, fc1w_h, HID * DIM);
    cvt_f32_f16_kernel<<<(DIM * HID + 255) / 256, 256, 0, stream>>>(fc2_w, fc2w_h, DIM * HID);
    gather_xw_kernel<<<(TOK * DIM) / 256, 256, 0, stream>>>(x, xw_h);
    cpb_kernel<<<2197, 128, 0, stream>>>(cpb_w1, cpb_b1, cpb_w2, tbl);

    // QKV: [21952,384] x [1152,384]^T
    gemm_wmma_kernel<<<dim3(TOK / 64, (3 * DIM) / 64), 128, 0, stream>>>(
        xw_h, qkvw_h, qkvb, qkv_o, nullptr, TOK, DIM, 3 * DIM, 0);

    // attention (flash) per window-head
    attn_kernel<<<BATCH * NW * HEADS, 256, 0, stream>>>(qkv_o, lscale, tbl, attn_h);

    // proj: [21952,384] x [384,384]^T
    gemm_wmma_kernel<<<dim3(TOK / 64, DIM / 64), 128, 0, stream>>>(
        attn_h, projw_h, proj_b, proj_o, nullptr, TOK, DIM, DIM, 0);

    // x1 = x + LN(window_reverse(roll(proj_out)))
    ln_res_kernel<<<BATCH * L, 128, 0, stream>>>(proj_o, x, n1g, n1b, x1_f, x1_h, 1);

    // MLP
    gemm_wmma_kernel<<<dim3(TOK / 64, HID / 64), 128, 0, stream>>>(
        x1_h, fc1w_h, fc1_b, nullptr, h_h, TOK, DIM, HID, 1);
    gemm_wmma_kernel<<<dim3(TOK / 64, DIM / 64), 128, 0, stream>>>(
        h_h, fc2w_h, fc2_b, y_f, nullptr, TOK, HID, DIM, 0);

    // out = x1 + LN(y)
    ln_res_kernel<<<BATCH * L, 128, 0, stream>>>(y_f, x1_f, n2g, n2b, out, nullptr, 0);
    (void)in_sizes; (void)n_in; (void)out_size; (void)ws_size;
}
